// Multi_Head_Attention_412316860838
// MI455X (gfx1250) — matrix operand order repaired and checked by reading, not compiled, not run on hardware
//
#include <hip/hip_runtime.h>
#include <stdint.h>

#define HIDDEN 1024
#define HEADS 16
#define HEAD_SIZE 64
#define SEQ 2048
#define BATCH 2

typedef __attribute__((ext_vector_type(16))) __bf16 v16bf;
typedef __attribute__((ext_vector_type(8)))  float  v8f;

union AFrag { v16bf v; unsigned u[8]; };
union CFrag { v8f   v; float    f[8]; };

__device__ __forceinline__ unsigned short f2bf(float f) {
  union { __bf16 h; unsigned short u; } r; r.h = (__bf16)f; return r.u;
}
__device__ __forceinline__ unsigned pack2bf(float lo, float hi) {
  union { __bf16 h[2]; unsigned u; } r;
  r.h[0] = (__bf16)lo; r.h[1] = (__bf16)hi;
  return r.u;
}

// async 16B global -> LDS copy (CDNA5 GLOBAL_LOAD_ASYNC_TO_LDS_B128, ASYNCcnt-tracked)
__device__ __forceinline__ void async_copy_b128(unsigned lds_off, const void* g) {
  asm volatile("global_load_async_to_lds_b128 %0, %1, off"
               :: "v"(lds_off), "v"((unsigned long long)(uintptr_t)g) : "memory");
}

// ---------------- fp32 -> bf16 pair conversion (row-major, pairs along last dim) -----
__global__ void __launch_bounds__(256) cvt_f32_bf16(const float* __restrict__ x,
                                                    unsigned* __restrict__ out, int nwords) {
  int i = blockIdx.x * blockDim.x + threadIdx.x;
  if (i < nwords) out[i] = pack2bf(x[2 * i], x[2 * i + 1]);
}

// ---------------- pack torch-style W [N][K] into WMMA B-fragment order ----------------
// Bw[(((k/32)*(N/16)) + ntile)*256 + lane*8 + i] = pack(W[col][k], W[col][k+1])
// with lane = n + 16*h2, col = ntile*16 + n, k = kb*32 + 16*h2 + 2*i  (B[kdim][n]=W[n][kdim])
__global__ void __launch_bounds__(256) pack_weight_frag(const float* __restrict__ W,
                                                        unsigned* __restrict__ Bw,
                                                        int N, int K) {
  int idx = blockIdx.x * blockDim.x + threadIdx.x;     // (K/32)*(N/16)*256 words
  int i     = idx & 7;
  int lane  = (idx >> 3) & 31;
  int tile  = idx >> 8;
  int ntile = tile % (N >> 4);
  int kb    = tile / (N >> 4);
  int n = lane & 15, h2 = lane >> 4;
  int k   = kb * 32 + 16 * h2 + 2 * i;
  int col = ntile * 16 + n;
  Bw[idx] = pack2bf(W[(size_t)col * K + k], W[(size_t)col * K + k + 1]);
}

// ---------------- generic bf16 WMMA GEMM: C[M,N] = A[M,K] * Bw ------------------------
// A32: bf16 row-major as dwords (pairs along K). Bw: fragment-packed (above).
// mode 0: out bf16 row-major; mode 1: out bf16 in V-fragment order (for attention);
// mode 2: out fp32 row-major.
__global__ void __launch_bounds__(128) gemm_bf16(const unsigned* __restrict__ A32,
                                                 const unsigned* __restrict__ Bw,
                                                 void* __restrict__ outp,
                                                 int M, int N, int K, int mode) {
  const int lane = threadIdx.x & 31;
  const int wave = threadIdx.x >> 5;
  const int wg   = blockIdx.x * 4 + wave;
  const int tiles_n = N >> 6;
  const int mt = wg / tiles_n;
  const int nb = wg % tiles_n;
  const int n  = lane & 15;
  const int h2 = lane >> 4;
  const int K2 = K >> 1;
  const int nt16 = N >> 4;

  CFrag acc[4];
#pragma unroll
  for (int t = 0; t < 4; ++t)
#pragma unroll
    for (int r = 0; r < 8; ++r) acc[t].f[r] = 0.f;

  const unsigned* Arow = A32 + (size_t)(mt * 16 + n) * K2;
  for (int kc = 0; kc < K; kc += 32) {
    AFrag a;
#pragma unroll
    for (int i = 0; i < 8; ++i)   // A pairs: k = kc + 16*h2 + 2*i
      a.u[i] = Arow[(kc >> 1) + 8 * h2 + i];
    const unsigned* Bbase = Bw + ((size_t)(kc >> 5) * nt16 + nb * 4) * 256 + lane * 8;
    __builtin_prefetch(Arow + (kc >> 1) + 16, 0, 1);
    __builtin_prefetch(Bbase + (size_t)nt16 * 256, 0, 1);
#pragma unroll
    for (int nt = 0; nt < 4; ++nt) {
      AFrag b;
#pragma unroll
      for (int i = 0; i < 8; ++i)  // contiguous -> 2x global_load_b128
        b.u[i] = Bbase[nt * 256 + i];
      acc[nt].v = __builtin_amdgcn_wmma_f32_16x16x32_bf16(
          false, a.v, false, b.v, (short)0, acc[nt].v, false, false);
    }
  }

  const int col = nb * 64;
  if (mode == 0) {
    unsigned short* o = (unsigned short*)outp;
#pragma unroll
    for (int nt = 0; nt < 4; ++nt)
#pragma unroll
      for (int r = 0; r < 8; ++r)
        o[(size_t)(mt * 16 + r + 8 * h2) * N + col + nt * 16 + n] = f2bf(acc[nt].f[r]);
  } else if (mode == 1) {
    // scatter into V-fragment order:
    // Vw[(((b*64 + kb)*64) + ntile)*256 + (n + 16*h2v)*8 + i]
    unsigned* o = (unsigned*)outp;
#pragma unroll
    for (int nt = 0; nt < 4; ++nt)
#pragma unroll
      for (int j = 0; j < 4; ++j) {
        int m0 = mt * 16 + 8 * h2 + 2 * j;      // even token index
        int bb = m0 >> 11;
        int s  = m0 & (SEQ - 1);
        int kb = s >> 5;
        int kk = s & 31;
        int h2v = kk >> 4;
        int i   = (kk >> 1) & 7;
        int ntile = nb * 4 + nt;
        o[(((size_t)bb * 64 + kb) * 64 + ntile) * 256 + (n + 16 * h2v) * 8 + i] =
            pack2bf(acc[nt].f[2 * j], acc[nt].f[2 * j + 1]);
      }
  } else {
    float* o = (float*)outp;
#pragma unroll
    for (int nt = 0; nt < 4; ++nt)
#pragma unroll
      for (int r = 0; r < 8; ++r)
        o[(size_t)(mt * 16 + r + 8 * h2) * N + col + nt * 16 + n] = acc[nt].f[r];
  }
}

// ---------------- fused attention: raw scores -> d_out attn region; raw@V -> ab -------
// 4 waves/block share (b,h): K/V blocks staged once per block via async global->LDS,
// double-buffered (copy of block kb+1 overlaps WMMAs of block kb).
__global__ void __launch_bounds__(128) attn_raw_kernel(const unsigned* __restrict__ Q32,
                                                       const unsigned* __restrict__ K32,
                                                       const unsigned* __restrict__ Vw,
                                                       float* __restrict__ attn,
                                                       unsigned short* __restrict__ ab) {
  __shared__ unsigned stage[2][2048];          // [buf][ 1024 K dwords | 1024 V dwords ]
  __shared__ unsigned short sc[4][16 * 32];    // per-wave 16q x 32k bf16 score staging
  const int lane = threadIdx.x & 31;
  const int wave = threadIdx.x >> 5;
  const int wg   = blockIdx.x * 4 + wave;      // 4096 waves total
  const int qt = wg & 127;
  const int h  = (wg >> 7) & 15;
  const int b  = wg >> 11;
  const int n  = lane & 15;
  const int h2 = lane >> 4;

  // Q A-fragments (channels 0..31 and 32..63); row stride = 512 dwords
  const int qtok = b * SEQ + qt * 16 + n;
  const unsigned* Qrow = Q32 + (size_t)qtok * 512 + h * 32;
  AFrag qa[2];
#pragma unroll
  for (int c = 0; c < 2; ++c)
#pragma unroll
    for (int i = 0; i < 8; ++i)
      qa[c].u[i] = Qrow[c * 16 + 8 * h2 + i];

  CFrag oacc[4];
#pragma unroll
  for (int t = 0; t < 4; ++t)
#pragma unroll
    for (int r = 0; r < 8; ++r) oacc[t].f[r] = 0.f;

  unsigned short* my = sc[wave];
  float* attnbase = attn + (size_t)(b * HEADS + h) * SEQ * SEQ;

  // ---- async stage issue: 8KB (4 b128 per thread) for key-block kb2 into stage[buf]
  auto issue_stage = [&](int buf, int kb2) {
    unsigned base = (unsigned)(uintptr_t)&stage[buf][0];
    const unsigned* Ksrc = K32 + (size_t)(b * SEQ + kb2 * 32) * 512 + h * 32;
    const unsigned* Vsrc = Vw + (((size_t)b * 64 + kb2) * 64 + h * 4) * 256;
    int tid = threadIdx.x;
#pragma unroll
    for (int c = 0; c < 4; ++c) {
      int d = (tid + c * 128) * 4;             // dword index in [0,2048)
      const unsigned* src;
      if (d < 1024) {                          // K region: 32 tokens x 32 dwords
        src = Ksrc + (size_t)(d >> 5) * 512 + (d & 31);
      } else {                                 // V region: fragment-linear
        src = Vsrc + (d - 1024);
      }
      async_copy_b128(base + (unsigned)d * 4u, src);
    }
  };

  issue_stage(0, 0);
  for (int kb = 0; kb < SEQ / 32; ++kb) {
    const int cur = kb & 1;
    if (kb < SEQ / 32 - 1) {
      issue_stage(cur ^ 1, kb + 1);            // overlap next copy with this compute
      asm volatile("s_wait_asynccnt 0x4" ::: "memory");  // prev 4 done, new 4 in flight
    } else {
      asm volatile("s_wait_asynccnt 0x0" ::: "memory");
    }
    __syncthreads();
    const unsigned* Kl = &stage[cur][0];
    const unsigned* Vl = &stage[cur][1024];

    // --- raw score tiles (16q x 32k), K fragments from LDS ---
#pragma unroll
    for (int t = 0; t < 2; ++t) {
      CFrag sacc;
#pragma unroll
      for (int r = 0; r < 8; ++r) sacc.f[r] = 0.f;
#pragma unroll
      for (int c = 0; c < 2; ++c) {
        AFrag kf;
#pragma unroll
        for (int i = 0; i < 8; ++i)            // contiguous -> ds_load_b128
          kf.u[i] = Kl[(t * 16 + n) * 32 + c * 16 + 8 * h2 + i];
        sacc.v = __builtin_amdgcn_wmma_f32_16x16x32_bf16(
            false, qa[c].v, false, kf.v, (short)0, sacc.v, false, false);
      }
      // write raw scores (fp32) to output attn region + stage bf16 tile in LDS
#pragma unroll
      for (int r = 0; r < 8; ++r) {
        int qrow = qt * 16 + r + 8 * h2;
        attnbase[(size_t)qrow * SEQ + kb * 32 + t * 16 + n] = sacc.f[r];
        my[(r + 8 * h2) * 32 + t * 16 + n] = f2bf(sacc.f[r]);
      }
    }
    asm volatile("s_wait_dscnt 0x0" ::: "memory");
    // reload score tile as A-fragment (m=q, kdim=key within 32-block)
    AFrag sa;
#pragma unroll
    for (int i = 0; i < 8; ++i)
      sa.u[i] = ((const unsigned*)my)[n * 16 + 8 * h2 + i];
    // V B-fragments from LDS (fragment-packed: contiguous per lane), accumulate raw@V
#pragma unroll
    for (int ct = 0; ct < 4; ++ct) {
      AFrag vf;
#pragma unroll
      for (int i = 0; i < 8; ++i)              // contiguous -> ds_load_b128
        vf.u[i] = Vl[ct * 256 + lane * 8 + i];
      oacc[ct].v = __builtin_amdgcn_wmma_f32_16x16x32_bf16(
          false, sa.v, false, vf.v, (short)0, oacc[ct].v, false, false);
    }
    __syncthreads();                           // all waves done reading stage[cur]
  }

  // write attention output (heads concatenated, bf16 row-major) for final projection
#pragma unroll
  for (int ct = 0; ct < 4; ++ct)
#pragma unroll
    for (int r = 0; r < 8; ++r)
      ab[(size_t)(b * SEQ + qt * 16 + r + 8 * h2) * HIDDEN + h * 64 + ct * 16 + n] =
          f2bf(oacc[ct].f[r]);
}

// ---------------- in-place masked/scaled softmax over each 2048-wide row --------------
__global__ void __launch_bounds__(256) softmax_rows(float* __restrict__ attn,
                                                    const int* __restrict__ causal_p) {
  const int row = blockIdx.x;              // B*H*S rows
  const int q   = row & (SEQ - 1);
  const int tid = threadIdx.x;
  const int causal = causal_p[0];
  float* p = attn + (size_t)row * SEQ;

  float v[8];
  float mx = -__builtin_inff();
#pragma unroll
  for (int j = 0; j < 8; ++j) {
    int k = tid + j * 256;
    float s = p[k] * 0.125f;               // 1/sqrt(64)
    if (causal && k > q) s = -__builtin_inff();
    v[j] = s;
    mx = fmaxf(mx, s);
  }
  __shared__ float red[8];
  for (int m = 16; m > 0; m >>= 1) mx = fmaxf(mx, __shfl_xor(mx, m, 32));
  if ((tid & 31) == 0) red[tid >> 5] = mx;
  __syncthreads();
  float bmax = red[0];
#pragma unroll
  for (int j = 1; j < 8; ++j) bmax = fmaxf(bmax, red[j]);
  __syncthreads();

  float sum = 0.f;
#pragma unroll
  for (int j = 0; j < 8; ++j) { float e = __expf(v[j] - bmax); v[j] = e; sum += e; }
  for (int m = 16; m > 0; m >>= 1) sum += __shfl_xor(sum, m, 32);
  if ((tid & 31) == 0) red[tid >> 5] = sum;
  __syncthreads();
  float bsum = 0.f;
#pragma unroll
  for (int j = 0; j < 8; ++j) bsum += red[j];
  float inv = 1.0f / bsum;
#pragma unroll
  for (int j = 0; j < 8; ++j) p[tid + j * 256] = v[j] * inv;
}

// -------------------------------------------------------------------------------------
extern "C" void kernel_launch(void* const* d_in, const int* in_sizes, int n_in,
                              void* d_out, int out_size, void* d_ws, size_t ws_size,
                              hipStream_t stream) {
  const float* x      = (const float*)d_in[0];
  const float* Wq     = (const float*)d_in[1];
  const float* Wk     = (const float*)d_in[2];
  const float* Wv     = (const float*)d_in[3];
  const float* Wo     = (const float*)d_in[4];
  const int*   causal = (const int*)d_in[5];

  char* ws = (char*)d_ws;                                  // needs ~48 MB
  unsigned*       xb  = (unsigned*)(ws);                   //  8 MB: x bf16 pairs
  unsigned*       Wqp = (unsigned*)(ws + ((size_t)8  << 20));
  unsigned*       Wkp = (unsigned*)(ws + ((size_t)10 << 20));
  unsigned*       Wvp = (unsigned*)(ws + ((size_t)12 << 20));
  unsigned*       Wop = (unsigned*)(ws + ((size_t)14 << 20));
  unsigned*       Qb  = (unsigned*)(ws + ((size_t)16 << 20)); // 8 MB bf16 rowmajor
  unsigned*       Kb  = (unsigned*)(ws + ((size_t)24 << 20)); // 8 MB bf16 rowmajor
  unsigned*       Vw  = (unsigned*)(ws + ((size_t)32 << 20)); // 8 MB bf16 frag-packed
  unsigned short* ab  = (unsigned short*)(ws + ((size_t)40 << 20)); // 8 MB

  float* result = (float*)d_out;
  float* attn   = result + (size_t)BATCH * SEQ * HIDDEN;

  const int M = BATCH * SEQ;                               // 4096

  // 1) convert x to bf16 pairs
  cvt_f32_bf16<<<(M * HIDDEN / 2) / 256, 256, 0, stream>>>(x, xb, M * HIDDEN / 2);
  // 2) pack weights into WMMA B-fragment order
  const int wwords = (HIDDEN / 32) * (HIDDEN / 16) * 256;
  pack_weight_frag<<<wwords / 256, 256, 0, stream>>>(Wq, Wqp, HIDDEN, HIDDEN);
  pack_weight_frag<<<wwords / 256, 256, 0, stream>>>(Wk, Wkp, HIDDEN, HIDDEN);
  pack_weight_frag<<<wwords / 256, 256, 0, stream>>>(Wv, Wvp, HIDDEN, HIDDEN);
  pack_weight_frag<<<wwords / 256, 256, 0, stream>>>(Wo, Wop, HIDDEN, HIDDEN);
  // 3) QKV projections
  dim3 ggrid((M / 16) * (HIDDEN / 64) / 4);
  gemm_bf16<<<ggrid, 128, 0, stream>>>(xb, Wqp, Qb, M, HIDDEN, HIDDEN, 0);
  gemm_bf16<<<ggrid, 128, 0, stream>>>(xb, Wkp, Kb, M, HIDDEN, HIDDEN, 0);
  gemm_bf16<<<ggrid, 128, 0, stream>>>(xb, Wvp, Vw, M, HIDDEN, HIDDEN, 1);
  // 4) fused raw-score attention (writes raw scores into d_out attn region + ab)
  attn_raw_kernel<<<BATCH * HEADS * (SEQ / 16) / 4, 128, 0, stream>>>(Qb, Kb, Vw, attn, ab);
  // 5) in-place softmax over attn region -> attn_values output
  softmax_rows<<<BATCH * HEADS * SEQ, 256, 0, stream>>>(attn, causal);
  // 6) output projection -> fp32 result
  gemm_bf16<<<ggrid, 128, 0, stream>>>((const unsigned*)ab, Wop, result, M, HIDDEN, HIDDEN, 2);
}
